// GCN_26903675142382
// MI455X (gfx1250) — matrix operand order repaired and checked by reading, not compiled, not run on hardware
//
#include <hip/hip_runtime.h>

typedef __attribute__((ext_vector_type(16))) _Float16 v16h;
typedef __attribute__((ext_vector_type(8)))  float    v8f;

// ---------- helpers ----------
__device__ __forceinline__ unsigned enc_f32(float f) {
    unsigned b = __float_as_uint(f);
    return (b & 0x80000000u) ? ~b : (b | 0x80000000u);   // monotone map float -> unsigned
}
__device__ __forceinline__ float dec_f32(unsigned u) {
    unsigned b = (u & 0x80000000u) ? (u & 0x7fffffffu) : ~u;
    return __uint_as_float(b);
}

// ---------- degree / norm ----------
__global__ void deg_kernel(const int* __restrict__ dst, float* __restrict__ deg, int E) {
    int e = blockIdx.x * blockDim.x + threadIdx.x;
    if (e < E) atomicAdd(&deg[dst[e]], 1.0f);
}
__global__ void dinv_kernel(const float* __restrict__ deg, float* __restrict__ dinv, int N) {
    int n = blockIdx.x * blockDim.x + threadIdx.x;
    if (n < N) dinv[n] = rsqrtf(deg[n] + 1.0f);          // +1 = self loop; always > 0
}

// ---------- pack weight K x ncols (row-major f32) into WMMA B-fragment order (f16) ----------
// B 32x16 f16 order used here (contiguous k, not the ISA table order): lane holds column n = lane%16;
// K = kt*32 + (lane<16 ? 0 : 16) + j, element j of v16h (VGPR j/2, half j%2).
__global__ void pack_w(const float* __restrict__ W, _Float16* __restrict__ Bf, int K, int ncols) {
    int idx = blockIdx.x * blockDim.x + threadIdx.x;
    int total = K * ncols;
    if (idx >= total) return;
    int j    = idx & 15;
    int lane = (idx >> 4) & 31;
    int t    = idx >> 9;              // tile = kt*ntn + tn (512 elems per tile)
    int ntn  = ncols >> 4;
    int kt   = t / ntn;
    int tn   = t - kt * ntn;
    int k    = (kt << 5) + ((lane >> 4) << 4) + j;
    int n    = (tn << 4) + (lane & 15);
    Bf[idx] = (_Float16)W[k * ncols + n];
}

// ---------- WMMA GEMM: C[nRows x NTN*16] = [A | A2][nRows x NKT*32] (f32) @ Bfrag (f16) ----------
// Weight fragments (NKT*NTN KB, <=32KB) are staged in LDS once per workgroup; all waves
// share them. One wave computes a full 16-row x ncols stripe: A fragment loaded once per
// K-chunk, reused across NTN accumulators (back-to-back WMMAs, distinct D -> no hazards).
// A 16x32 f16 order used here (contiguous k, same as B packing): row m = lane%16;
//   elements 0..7  : K = (lane<16 ? 0 : 16)  + j
//   elements 8..15 : K = (lane<16 ? 8 : 24) + (j-8)
template <int NKT, int KSPLIT_KT, int NTN>
__global__ void gemm_wmma(const float* __restrict__ A,  int sA,
                          const float* __restrict__ A2, int sA2,
                          const _Float16* __restrict__ Bfrag,
                          float* __restrict__ C, int nRows) {
    constexpr int FRAG_ELEMS = NKT * NTN * 512;          // f16 elements in all fragments
    __shared__ __align__(32) _Float16 Bs[FRAG_ELEMS];

    // cooperative stage: global -> LDS (all threads participate, before any early-out)
    {
        const uint4* g = (const uint4*)Bfrag;
        uint4*       l = (uint4*)Bs;
        constexpr int CHUNKS = FRAG_ELEMS / 8;           // 16B chunks
        for (int i = threadIdx.x; i < CHUNKS; i += blockDim.x) l[i] = g[i];
    }
    __syncthreads();

    const int lane = threadIdx.x & 31;
    const int tm   = blockIdx.x * (blockDim.x >> 5) + (threadIdx.x >> 5);
    if (tm * 16 >= nRows) return;                 // wave-uniform: EXEC stays all-1s
    const int hi   = lane >> 4;                   // 0 or 1
    const int row  = tm * 16 + (lane & 15);
    const int sel0 = hi ? 16 : 0;
    const int sel1 = hi ? 24 : 8;
    constexpr int ncols = NTN * 16;

    v8f acc[NTN];
#pragma unroll
    for (int tn = 0; tn < NTN; ++tn) acc[tn] = v8f{};

#pragma unroll
    for (int kt = 0; kt < NKT; ++kt) {
        const float* src = (kt < KSPLIT_KT) ? A : A2;
        const int    s   = (kt < KSPLIT_KT) ? sA : sA2;
        const int    kb  = (kt < KSPLIT_KT) ? kt * 32 : (kt - KSPLIT_KT) * 32;
        const long   rb  = (long)row * s + kb;
        const float4 f0 = *(const float4*)(src + rb + sel0);
        const float4 f1 = *(const float4*)(src + rb + sel0 + 4);
        const float4 f2 = *(const float4*)(src + rb + sel1);
        const float4 f3 = *(const float4*)(src + rb + sel1 + 4);
        v16h a;
        a[0]=(_Float16)f0.x;  a[1]=(_Float16)f0.y;  a[2]=(_Float16)f0.z;  a[3]=(_Float16)f0.w;
        a[4]=(_Float16)f1.x;  a[5]=(_Float16)f1.y;  a[6]=(_Float16)f1.z;  a[7]=(_Float16)f1.w;
        a[8]=(_Float16)f2.x;  a[9]=(_Float16)f2.y;  a[10]=(_Float16)f2.z; a[11]=(_Float16)f2.w;
        a[12]=(_Float16)f3.x; a[13]=(_Float16)f3.y; a[14]=(_Float16)f3.z; a[15]=(_Float16)f3.w;

#pragma unroll
        for (int tn = 0; tn < NTN; ++tn) {
            v16h b = *(const v16h*)(Bs + ((kt * NTN + tn) * 32 + lane) * 16);
            acc[tn] = __builtin_amdgcn_wmma_f32_16x16x32_f16(false, a, false, b,
                                                             (short)0, acc[tn], false, false);
        }
    }
    // D layout: VGPR r -> M = tm*16 + r + (hi?8:0), N = tn*16 + lane%16
    const int nlo   = lane & 15;
    const int mrow0 = tm * 16 + (hi ? 8 : 0);
#pragma unroll
    for (int tn = 0; tn < NTN; ++tn) {
        const int col = tn * 16 + nlo;
#pragma unroll
        for (int r = 0; r < 8; ++r)
            C[(long)(mrow0 + r) * ncols + col] = acc[tn][r];
    }
}

// ---------- edge scatter: agg[dst] += lin[src] * dinv[src]*dinv[dst] ----------
__global__ void scatter_edges(const int* __restrict__ src, const int* __restrict__ dst,
                              const float* __restrict__ dinv, const float* __restrict__ lin,
                              float* __restrict__ agg, int E, int F, int lpeShift) {
    int t = blockIdx.x * blockDim.x + threadIdx.x;
    int e = t >> lpeShift;                       // F/4 lanes per edge
    if (e >= E) return;
    int l = t - (e << lpeShift);
    int s = src[e], d = dst[e];
    float nrm = dinv[s] * dinv[d];
    const float4 v = *(const float4*)(lin + (long)s * F + l * 4);
    float* o = agg + (long)d * F + l * 4;
    atomicAdd(o + 0, v.x * nrm);
    atomicAdd(o + 1, v.y * nrm);
    atomicAdd(o + 2, v.z * nrm);
    atomicAdd(o + 3, v.w * nrm);
}

// ---------- self-loop + bias + optional relu (runs after scatter; stream-serialized) ----------
__global__ void selfloop_bias_act(float* __restrict__ agg, const float* __restrict__ lin,
                                  const float* __restrict__ dinv, const float* __restrict__ bias,
                                  int N, int F, int relu) {
    long t = (long)blockIdx.x * blockDim.x + threadIdx.x;
    if (t >= (long)N * F) return;
    int n = (int)(t / F);
    int c = (int)(t - (long)n * F);
    float di = dinv[n];
    float v = agg[t] + lin[t] * di * di + bias[c];
    agg[t] = relu ? fmaxf(v, 0.0f) : v;
}

// ---------- gate score + segment max ----------
__global__ void gate_kernel(const float* __restrict__ h3, const float* __restrict__ gW,
                            const float* __restrict__ gB, const int* __restrict__ batch,
                            float* __restrict__ gate, unsigned* __restrict__ gmax, int N) {
    int lane = threadIdx.x & 31;
    int n = blockIdx.x * (blockDim.x >> 5) + (threadIdx.x >> 5);
    if (n >= N) return;
    float p = h3[(long)n * 64 + lane]      * gW[lane]
            + h3[(long)n * 64 + lane + 32] * gW[lane + 32];
#pragma unroll
    for (int off = 16; off > 0; off >>= 1) p += __shfl_xor(p, off, 32);
    if (lane == 0) {
        float g = p + gB[0];
        gate[n] = g;
        atomicMax(&gmax[batch[n]], enc_f32(g));
    }
}

// ---------- exp(gate - max) + segment sum ----------
__global__ void expdenom_kernel(float* __restrict__ gate, const unsigned* __restrict__ gmax,
                                const int* __restrict__ batch, float* __restrict__ denom, int N) {
    int n = blockIdx.x * blockDim.x + threadIdx.x;
    if (n >= N) return;
    int b = batch[n];
    float e = expf(gate[n] - dec_f32(gmax[b]));
    gate[n] = e;
    atomicAdd(&denom[b], e);
}

// ---------- weighted pooling ----------
__global__ void pool_kernel(const float* __restrict__ h3, const float* __restrict__ e,
                            const float* __restrict__ denom, const int* __restrict__ batch,
                            float* __restrict__ pooled, int N) {
    long t = (long)blockIdx.x * blockDim.x + threadIdx.x;
    if (t >= (long)N * 64) return;
    int n = (int)(t >> 6);
    int c = (int)(t & 63);
    int b = batch[n];
    float alpha = e[n] / denom[b];
    atomicAdd(&pooled[b * 64 + c], alpha * h3[(long)n * 64 + c]);
}

// ---------- log_softmax over [G,64] ----------
__global__ void logsoftmax64(const float* __restrict__ pooled, float* __restrict__ out) {
    __shared__ float red[64];
    int g = blockIdx.x, c = threadIdx.x;
    float v = pooled[g * 64 + c];
    red[c] = v; __syncthreads();
    for (int s = 32; s > 0; s >>= 1) { if (c < s) red[c] = fmaxf(red[c], red[c + s]); __syncthreads(); }
    float mx = red[0]; __syncthreads();
    red[c] = expf(v - mx); __syncthreads();
    for (int s = 32; s > 0; s >>= 1) { if (c < s) red[c] += red[c + s]; __syncthreads(); }
    float lse = logf(red[0]) + mx;
    out[g * 64 + c] = v - lse;
}

// ================= host side =================
static inline int ceil_div(long a, int b) { return (int)((a + b - 1) / b); }

extern "C" void kernel_launch(void* const* d_in, const int* in_sizes, int n_in,
                              void* d_out, int out_size, void* d_ws, size_t ws_size,
                              hipStream_t stream) {
    const int IN = 128, H = 128, OUT = 64;
    const int N = in_sizes[0] / IN;
    const int E = in_sizes[1] / 2;
    const int G = out_size / OUT;

    const float* x      = (const float*)d_in[0];
    const int*   ei     = (const int*)  d_in[1];
    const int*   src    = ei;
    const int*   dst    = ei + E;
    const int*   batch  = (const int*)  d_in[2];
    const float* W1     = (const float*)d_in[3];
    const float* b1     = (const float*)d_in[4];
    const float* W2     = (const float*)d_in[5];
    const float* b2     = (const float*)d_in[6];
    const float* W3     = (const float*)d_in[7];
    const float* b3     = (const float*)d_in[8];
    const float* gW     = (const float*)d_in[9];
    const float* gB     = (const float*)d_in[10];
    float*       out    = (float*)d_out;

    // carve workspace (256B aligned)
    char* base = (char*)d_ws;
    size_t off = 0;
    auto carve = [&](size_t bytes) -> char* {
        char* p = base + off;
        off = (off + bytes + 255) & ~(size_t)255;
        return p;
    };
    float*     lin    = (float*)    carve((size_t)N * H * 4);
    float*     agg    = (float*)    carve((size_t)N * H * 4);
    _Float16*  w1f    = (_Float16*) carve((size_t)IN * H * 2);
    _Float16*  w2f    = (_Float16*) carve((size_t)H  * H * 2);
    _Float16*  w3f    = (_Float16*) carve((size_t)(H + IN) * OUT * 2);
    float*     degb   = (float*)    carve((size_t)N * 4);
    float*     dinvb  = (float*)    carve((size_t)N * 4);
    float*     gateb  = (float*)    carve((size_t)N * 4);
    unsigned*  gmaxb  = (unsigned*) carve((size_t)G * 4);
    float*     denomb = (float*)    carve((size_t)G * 4);
    float*     pooled = (float*)    carve((size_t)G * OUT * 4);
    (void)ws_size;

    const int B = 256;

    // degrees + normalization
    hipMemsetAsync(degb, 0, (size_t)N * 4, stream);
    deg_kernel<<<ceil_div(E, B), B, 0, stream>>>(dst, degb, E);
    dinv_kernel<<<ceil_div(N, B), B, 0, stream>>>(degb, dinvb, N);

    // pack weights into WMMA B-fragment layout
    pack_w<<<ceil_div((long)IN * H, B), B, 0, stream>>>(W1, w1f, IN, H);
    pack_w<<<ceil_div((long)H  * H, B), B, 0, stream>>>(W2, w2f, H, H);
    pack_w<<<ceil_div((long)(H + IN) * OUT, B), B, 0, stream>>>(W3, w3f, H + IN, OUT);

    const int tilesM = N / 16;                    // 50000/16 = 3125 exact
    const int gemmBlocks = ceil_div(tilesM, B / 32);

    // ---- layer 1: lin = x @ W1  (K=128, ncols=128) ----
    gemm_wmma<4, 4, 8><<<gemmBlocks, B, 0, stream>>>(x, IN, x, IN, w1f, lin, N);
    hipMemsetAsync(agg, 0, (size_t)N * H * 4, stream);
    scatter_edges<<<ceil_div((long)E * (H / 4), B), B, 0, stream>>>(src, dst, dinvb, lin, agg, E, H, 5);
    selfloop_bias_act<<<ceil_div((long)N * H, B), B, 0, stream>>>(agg, lin, dinvb, b1, N, H, 1);

    // ---- layer 2: lin = h1 @ W2  (K=128, ncols=128) ----
    gemm_wmma<4, 4, 8><<<gemmBlocks, B, 0, stream>>>(agg, H, agg, H, w2f, lin, N);
    hipMemsetAsync(agg, 0, (size_t)N * H * 4, stream);
    scatter_edges<<<ceil_div((long)E * (H / 4), B), B, 0, stream>>>(src, dst, dinvb, lin, agg, E, H, 5);
    selfloop_bias_act<<<ceil_div((long)N * H, B), B, 0, stream>>>(agg, lin, dinvb, b2, N, H, 1);

    // ---- layer 3: lin = [h2 | x] @ W3  (K=256 split at 128, ncols=64) ----
    gemm_wmma<8, 4, 4><<<gemmBlocks, B, 0, stream>>>(agg, H, x, IN, w3f, lin, N);
    hipMemsetAsync(agg, 0, (size_t)N * OUT * 4, stream);
    scatter_edges<<<ceil_div((long)E * (OUT / 4), B), B, 0, stream>>>(src, dst, dinvb, lin, agg, E, OUT, 4);
    selfloop_bias_act<<<ceil_div((long)N * OUT, B), B, 0, stream>>>(agg, lin, dinvb, b3, N, OUT, 0);

    // ---- global attention pooling ----
    hipMemsetAsync(gmaxb, 0, (size_t)G * 4, stream);      // enc-space minimum
    hipMemsetAsync(denomb, 0, (size_t)G * 4, stream);
    hipMemsetAsync(pooled, 0, (size_t)G * OUT * 4, stream);
    gate_kernel<<<ceil_div(N, B / 32), B, 0, stream>>>(agg, gW, gB, batch, gateb, gmaxb, N);
    expdenom_kernel<<<ceil_div(N, B), B, 0, stream>>>(gateb, gmaxb, batch, denomb, N);
    pool_kernel<<<ceil_div((long)N * OUT, B), B, 0, stream>>>(agg, gateb, denomb, batch, pooled, N);
    logsoftmax64<<<G, 64, 0, stream>>>(pooled, out);
}
